// OptimalTransportBridge_3530463117427
// MI455X (gfx1250) — matrix operand order repaired and checked by reading, not compiled, not run on hardware
//
// MI455X (gfx1250, wave32) Optimal-Transport-Bridge pipeline.
// Dominant GEMM (137 GFLOP, AI~490 FLOP/B -> compute bound) runs on
// v_wmma_f32_16x16x32_bf16 with bf16x3 (hi/lo split) for ~f32 accuracy at
// 8x the MACs/instruction of the f32 16x16x4 WMMA path. A-tiles of the big
// GEMM are DMA'd global->LDS with the Tensor Data Mover (double buffered,
// TENSORcnt-pipelined) so the DMA of tile t+1 overlaps the WMMAs of tile t.
//
// Workspace layout (floats), total ~147 MB (assumed <= ws_size):
//   src_proj[16384*2048] | Cmat[16384*64] | Kmat[16384*64] |
//   soft_raw[8*64*2048] | scale[2048] | anchS[64*2048] | anchSq[64] |
//   a[8*2048] | u[8*2048] | v[8*64] | colsum[8*64] | scal[8]

#include <hip/hip_runtime.h>
#include <stdint.h>

#define B_  8
#define S_  2048
#define D_  2048
#define T_  2048
#define L_  64
#define M_  (B_*S_)
#define EPS_INV 10.0f     // 1/0.1
#define N_ITERS 20

typedef __attribute__((ext_vector_type(16))) __bf16    v16bf;
typedef __attribute__((ext_vector_type(8)))  float     v8f;
typedef __attribute__((ext_vector_type(4)))  unsigned  v4u;
typedef __attribute__((ext_vector_type(4)))  int       v4i;
typedef __attribute__((ext_vector_type(8)))  int       v8i;

// ---------------- Tensor Data Mover (TDM) ----------------
#if defined(__has_builtin)
#if __has_builtin(__builtin_amdgcn_tensor_load_to_lds)
#define HAVE_TDM 1
#endif
#endif
#ifndef HAVE_TDM
#define HAVE_TDM 0
#endif

#if HAVE_TDM
// The 6-arg builtin ships with the toolchain that provides the TDM header;
// ROCm 7.2 (clang-22) has the 5-arg form and no header.
#if __has_include(<hip/amd_detail/amd_gfx1250_TDM.h>)
#define TDM_SIX_ARGS 1
#else
#define TDM_SIX_ARGS 0
#endif

// 2-D tile DMA: global (row-major, stride0 elements/row) -> LDS, f32 elements.
// D# group0: count=1 | lds_addr | global_addr | type=2("image")
// D# group1: data_size=4B, tensor_dim0/1, tile_dim0/1, tensor_dim0_stride.
static __device__ __forceinline__ void tdm_load_tile_f32(
        unsigned lds_off, const void* gaddr,
        unsigned tile_x, unsigned tile_y,
        unsigned tdim0, unsigned tdim1, unsigned stride0) {
    unsigned long long ga = (unsigned long long)(size_t)gaddr;
    v4u g0 = { 1u,                                    // count=1, user descriptor
               lds_off,                               // lds_addr (bytes)
               (unsigned)ga,                          // global_addr[31:0]
               ((unsigned)(ga >> 32) & 0x1FFFFFFu) | (2u << 30) };  // ga[56:32] | type=2
    v8i g1 = { (int)(2u << 16),                                        // data_size=4B
               (int)((tdim0 & 0xFFFFu) << 16),                         // tensor_dim0 lo
               (int)(((tdim0 >> 16) & 0xFFFFu) | ((tdim1 & 0xFFFFu) << 16)),
               (int)(((tdim1 >> 16) & 0xFFFFu) | ((tile_x & 0xFFFFu) << 16)),
               (int)(tile_y & 0xFFFFu),                                // tile_dim1 (tile_dim2=0)
               (int)stride0,                                           // tensor_dim0_stride lo
               0, 0 };
    v4i z4 = {0, 0, 0, 0};
#if TDM_SIX_ARGS
    v8i z8 = {0, 0, 0, 0, 0, 0, 0, 0};
    __builtin_amdgcn_tensor_load_to_lds(g0, g1, z4, z4, z8, 0);
#else
    __builtin_amdgcn_tensor_load_to_lds(g0, g1, z4, z4, 0);
#endif
}
#endif // HAVE_TDM

// ---------- bf16 split helpers ----------
static __device__ __forceinline__ unsigned short f32_to_bf16_rne(float x) {
    unsigned u = __float_as_uint(x);
    unsigned r = u + 0x7FFFu + ((u >> 16) & 1u);
    return (unsigned short)(r >> 16);
}
static __device__ __forceinline__ float bf16_bits_to_f32(unsigned short h) {
    return __uint_as_float(((unsigned)h) << 16);
}
static __device__ __forceinline__ void split_bf16(float x, unsigned short& hi, unsigned short& lo) {
    hi = f32_to_bf16_rne(x);
    lo = f32_to_bf16_rne(x - bf16_bits_to_f32(hi));
}

struct Bits256 { uint4 a, b; };
static __device__ __forceinline__ v16bf ld_frag(const unsigned short* p0, const unsigned short* p1) {
    Bits256 f;
    f.a = *(const uint4*)p0;
    f.b = *(const uint4*)p1;
    return __builtin_bit_cast(v16bf, f);
}
// D += A_hi*B_hi + A_hi*B_lo + A_lo*B_hi  (bf16x3 ~ f32 accuracy)
static __device__ __forceinline__ v8f wmma_bf16x3(v16bf ah, v16bf al, v16bf bh, v16bf bl, v8f c) {
    c = __builtin_amdgcn_wmma_f32_16x16x32_bf16(false, ah, false, bh, (short)0, c, false, false);
    c = __builtin_amdgcn_wmma_f32_16x16x32_bf16(false, ah, false, bl, (short)0, c, false, false);
    c = __builtin_amdgcn_wmma_f32_16x16x32_bf16(false, al, false, bh, (short)0, c, false, false);
    return c;
}

// ---------- small prep kernels ----------
__global__ __launch_bounds__(256) void k_scale_zero(const float* __restrict__ cost_scale,
                                                    float* __restrict__ scale,
                                                    float* __restrict__ scal) {
    int t = blockIdx.x * 256 + threadIdx.x;
    scale[t] = fabsf(cost_scale[t]) + 0.1f;
    if (blockIdx.x == 0 && threadIdx.x < 8) scal[threadIdx.x] = 0.f;
}

__global__ __launch_bounds__(256) void k_anchor(const float* __restrict__ anchors,
                                                const float* __restrict__ scale,
                                                float* __restrict__ anchS,
                                                float* __restrict__ anchSq) {
    int l = blockIdx.x, tid = threadIdx.x;
    __shared__ float red[256];
    float s = 0.f;
    for (int t = tid; t < T_; t += 256) {
        float v = anchors[(size_t)l * T_ + t] * scale[t];
        anchS[(size_t)l * T_ + t] = v;
        s += v * v;
    }
    red[tid] = s; __syncthreads();
    for (int off = 128; off; off >>= 1) { if (tid < off) red[tid] += red[tid + off]; __syncthreads(); }
    if (!tid) anchSq[l] = red[0];
}

__global__ __launch_bounds__(256) void k_avec(const int* __restrict__ mask, float* __restrict__ a) {
    int b = blockIdx.x, tid = threadIdx.x;
    __shared__ float red[256];
    __shared__ float inv;
    float s = 0.f;
    for (int i = tid; i < S_; i += 256) s += (float)mask[b * S_ + i];
    red[tid] = s; __syncthreads();
    for (int off = 128; off; off >>= 1) { if (tid < off) red[tid] += red[tid + off]; __syncthreads(); }
    if (!tid) inv = 1.f / (red[0] + 1e-8f);
    __syncthreads();
    for (int i = tid; i < S_; i += 256) a[b * S_ + i] = (float)mask[b * S_ + i] * inv;
}

// ---------- K1: src_proj = src_hidden @ W + b  (M=16384,N=2048,K=2048) ----------
#define PG_BM 64
#define PG_BN 128
#define PG_BK 32

__global__ __launch_bounds__(256)
void k_proj_gemm(const float* __restrict__ A, const float* __restrict__ W,
                 const float* __restrict__ bias, float* __restrict__ out) {
    __shared__ __align__(16) float          Af32[2][PG_BM][PG_BK];   // TDM stage (16KB)
    __shared__ __align__(16) unsigned short AsH[PG_BM][PG_BK];
    __shared__ __align__(16) unsigned short AsL[PG_BM][PG_BK];
    __shared__ __align__(16) unsigned short BsH[PG_BN][PG_BK]; // [n][k]
    __shared__ __align__(16) unsigned short BsL[PG_BN][PG_BK];

    const int tid = threadIdx.x, lane = tid & 31, wave = tid >> 5;
    const int wm = wave & 3, wn = wave >> 2;
    const int m0 = blockIdx.x * PG_BM, n0 = blockIdx.y * PG_BN;

    const int am  = lane & 15;
    const int akb = (lane >> 4) << 3;   // 0 or 8
    const int bn  = lane & 15;
    const int bkb = (lane >> 4) << 3;   // 0 or 8

    v8f acc[4] = {};
    int cur = 0;

#if HAVE_TDM
    // Prologue: DMA first A tile (64 rows x 32 k, f32) into buffer 0.
    if (tid == 0) {
        tdm_load_tile_f32((unsigned)(size_t)(void*)&Af32[0][0][0],
                          A + (size_t)m0 * D_,
                          PG_BK, PG_BM, (unsigned)D_, (unsigned)M_, (unsigned)D_);
        __builtin_amdgcn_s_wait_tensorcnt(0);
    }
    __syncthreads();
#endif

    for (int k0 = 0; k0 < D_; k0 += PG_BK) {
        // split staged A tile (f32 -> bf16 hi/lo) into WMMA-feed LDS arrays
        for (int idx = tid; idx < PG_BM * PG_BK; idx += 256) {
            int r = idx >> 5, c = idx & 31;
#if HAVE_TDM
            float x = Af32[cur][r][c];
#else
            float x = A[(size_t)(m0 + r) * D_ + (k0 + c)];
#endif
            unsigned short h, l;
            split_bf16(x, h, l);
            AsH[r][c] = h; AsL[r][c] = l;
        }
        // cooperative load + split of B tile 32x128 (stored transposed [n][k])
        for (int idx = tid; idx < PG_BK * PG_BN; idx += 256) {
            int r = idx >> 7, c = idx & 127;
            unsigned short h, l;
            split_bf16(W[(size_t)(k0 + r) * T_ + (n0 + c)], h, l);
            BsH[c][r] = h; BsL[c][r] = l;
        }
        __syncthreads();

        const bool more = (k0 + PG_BK) < D_;
#if HAVE_TDM
        // Kick the DMA for the NEXT A tile; it overlaps this tile's WMMAs.
        if (tid == 0 && more) {
            tdm_load_tile_f32((unsigned)(size_t)(void*)&Af32[cur ^ 1][0][0],
                              A + (size_t)m0 * D_ + (k0 + PG_BK),
                              PG_BK, PG_BM, (unsigned)D_, (unsigned)M_, (unsigned)D_);
        }
#endif
        if (more)
            __builtin_prefetch(&W[(size_t)(k0 + PG_BK + (tid & 7)) * T_ + n0], 0, 1);

        const unsigned short* arH = &AsH[wm * 16 + am][0];
        const unsigned short* arL = &AsL[wm * 16 + am][0];
        v16bf aH = ld_frag(arH + akb, arH + akb + 16);
        v16bf aL = ld_frag(arL + akb, arL + akb + 16);
        #pragma unroll
        for (int f = 0; f < 4; ++f) {
            const unsigned short* brH = &BsH[wn * 64 + f * 16 + bn][0];
            const unsigned short* brL = &BsL[wn * 64 + f * 16 + bn][0];
            v16bf bH = ld_frag(brH + bkb, brH + bkb + 16);
            v16bf bL = ld_frag(brL + bkb, brL + bkb + 16);
            acc[f] = wmma_bf16x3(aH, aL, bH, bL, acc[f]);
        }

#if HAVE_TDM
        if (tid == 0 && more)
            __builtin_amdgcn_s_wait_tensorcnt(0);   // drain next-tile DMA
#endif
        __syncthreads();
        cur ^= 1;
    }

    #pragma unroll
    for (int f = 0; f < 4; ++f) {
        int col = n0 + wn * 64 + f * 16 + (lane & 15);
        float bv = bias[col];
        #pragma unroll
        for (int v = 0; v < 8; ++v) {
            int row = m0 + wm * 16 + v + ((lane >> 4) << 3);
            out[(size_t)row * T_ + col] = acc[f][v] + bv;
        }
    }
}

// ---------- K2: cost / Gibbs kernel: C = |s|^2 + |a|^2 - 2 cross; K = exp(-C/eps) ----------
#define CK_BM 128
#define CK_BK 32

__global__ __launch_bounds__(256)
void k_cost(const float* __restrict__ proj, const float* __restrict__ scale,
            const float* __restrict__ anchS, const float* __restrict__ anchSq,
            float* __restrict__ Cmat, float* __restrict__ Kmat) {
    __shared__ __align__(16) unsigned short AsH[CK_BM][CK_BK];
    __shared__ __align__(16) unsigned short AsL[CK_BM][CK_BK];
    __shared__ __align__(16) unsigned short BsH[L_][CK_BK];
    __shared__ __align__(16) unsigned short BsL[L_][CK_BK];
    __shared__ float ssq[CK_BM];

    const int tid = threadIdx.x, lane = tid & 31, wave = tid >> 5;
    const int m0 = blockIdx.x * CK_BM;
    const int am  = lane & 15;
    const int akb = (lane >> 4) << 3;
    const int bn  = lane & 15;
    const int bkb = (lane >> 4) << 3;

    v8f acc[4] = {};
    float sqp[16];
    #pragma unroll
    for (int i = 0; i < 16; ++i) sqp[i] = 0.f;

    for (int k0 = 0; k0 < T_; k0 += CK_BK) {
        #pragma unroll
        for (int i = 0; i < 16; ++i) {                 // 128x32 A tile, static reg idx
            int idx = tid + 256 * i;
            int r = idx >> 5, c = idx & 31;
            float x = proj[(size_t)(m0 + r) * T_ + (k0 + c)] * scale[k0 + c];
            sqp[i] += x * x;
            unsigned short h, l; split_bf16(x, h, l);
            AsH[r][c] = h; AsL[r][c] = l;
        }
        for (int idx = tid; idx < L_ * CK_BK; idx += 256) {   // 64x32 anchor tile
            int l = idx >> 5, c = idx & 31;
            unsigned short h, lo;
            split_bf16(anchS[(size_t)l * T_ + (k0 + c)], h, lo);
            BsH[l][c] = h; BsL[l][c] = lo;
        }
        __syncthreads();
        if (k0 + CK_BK < T_)
            __builtin_prefetch(&proj[(size_t)(m0 + wave * 16) * T_ + (k0 + CK_BK)], 0, 1);
        const unsigned short* arH = &AsH[wave * 16 + am][0];
        const unsigned short* arL = &AsL[wave * 16 + am][0];
        v16bf aH = ld_frag(arH + akb, arH + akb + 16);
        v16bf aL = ld_frag(arL + akb, arL + akb + 16);
        #pragma unroll
        for (int f = 0; f < 4; ++f) {
            const unsigned short* brH = &BsH[f * 16 + bn][0];
            const unsigned short* brL = &BsL[f * 16 + bn][0];
            v16bf bH = ld_frag(brH + bkb, brH + bkb + 16);
            v16bf bL = ld_frag(brL + bkb, brL + bkb + 16);
            acc[f] = wmma_bf16x3(aH, aL, bH, bL, acc[f]);
        }
        __syncthreads();
    }
    // wave-level (wave32) reduction of per-row |src_scaled|^2
    #pragma unroll
    for (int i = 0; i < 16; ++i) {
        float v = sqp[i];
        for (int off = 16; off >= 1; off >>= 1) v += __shfl_xor(v, off, 32);
        if (lane == 0) ssq[wave + 8 * i] = v;
    }
    __syncthreads();
    #pragma unroll
    for (int f = 0; f < 4; ++f) {
        int l = f * 16 + (lane & 15);
        float asq = anchSq[l];
        #pragma unroll
        for (int v = 0; v < 8; ++v) {
            int row = wave * 16 + v + ((lane >> 4) << 3);
            float C = ssq[row] + asq - 2.f * acc[f][v];
            size_t idx = (size_t)(m0 + row) * L_ + l;
            Cmat[idx] = C;
            Kmat[idx] = __expf(-C * EPS_INV);
        }
    }
}

// ---------- K3: Sinkhorn, one workgroup per batch, u/v LDS-resident ----------
__global__ __launch_bounds__(256)
void k_sinkhorn(const float* __restrict__ Kmat, const float* __restrict__ a,
                float* __restrict__ u_g, float* __restrict__ v_g) {
    int b = blockIdx.x, tid = threadIdx.x;
    __shared__ float u_s[S_];
    __shared__ float v_s[L_];
    __shared__ float cp[256];
    const float* Kb = Kmat + (size_t)b * S_ * L_;
    for (int s = tid; s < S_; s += 256) u_s[s] = 1.f;
    __syncthreads();
    for (int it = 0; it < N_ITERS; ++it) {
        int l = tid >> 2, q = tid & 3;
        float part = 0.f;
        for (int s = q * 512; s < (q + 1) * 512; ++s)
            part += Kb[(size_t)s * L_ + l] * u_s[s];
        cp[tid] = part; __syncthreads();
        if (tid < L_) {
            float cs = cp[tid * 4] + cp[tid * 4 + 1] + cp[tid * 4 + 2] + cp[tid * 4 + 3];
            v_s[tid] = (1.f / (float)L_) / (cs + 1e-8f);
        }
        __syncthreads();
        for (int s = tid; s < S_; s += 256) {
            const float* kr = Kb + (size_t)s * L_;
            float rs = 0.f;
            #pragma unroll 8
            for (int l2 = 0; l2 < L_; ++l2) rs += kr[l2] * v_s[l2];
            u_s[s] = a[b * S_ + s] / (rs + 1e-8f);
        }
        __syncthreads();
    }
    for (int s = tid; s < S_; s += 256) u_g[b * S_ + s] = u_s[s];
    if (tid < L_) v_g[b * L_ + tid] = v_s[tid];
}

// ---------- K4b: column sums + OT cost + P statistics ----------
__global__ __launch_bounds__(256)
void k_pstats(const float* __restrict__ Kmat, const float* __restrict__ Cmat,
              const float* __restrict__ u_g, const float* __restrict__ v_g,
              float* __restrict__ colsum, float* __restrict__ scal) {
    int bl = blockIdx.x, b = bl >> 6, l = bl & 63, tid = threadIdx.x;
    const float* Kb = Kmat + (size_t)b * S_ * L_;
    const float* Cb = Cmat + (size_t)b * S_ * L_;
    float vl = v_g[bl];
    float cs = 0.f, pc = 0.f, p2 = 0.f;
    for (int s = tid; s < S_; s += 256) {
        float P = u_g[b * S_ + s] * Kb[(size_t)s * L_ + l] * vl;
        float C = Cb[(size_t)s * L_ + l];
        cs += P; pc += P * C; p2 += P * P;
    }
    __shared__ float r1[256], r2[256], r3[256];
    r1[tid] = cs; r2[tid] = pc; r3[tid] = p2; __syncthreads();
    for (int off = 128; off; off >>= 1) {
        if (tid < off) { r1[tid] += r1[tid + off]; r2[tid] += r2[tid + off]; r3[tid] += r3[tid + off]; }
        __syncthreads();
    }
    if (!tid) {
        colsum[bl] = r1[0];
        atomicAdd(&scal[0], r2[0]);   // sum P*C (all batches)
        atomicAdd(&scal[1], r1[0]);   // sum P
        atomicAdd(&scal[2], r3[0]);   // sum P^2
    }
}

// ---------- K4: soft_raw[b] = P^T @ src_proj  (per batch: 64 x 2048 x 2048) ----------
__global__ __launch_bounds__(256)
void k_soft_gemm(const float* __restrict__ Kmat, const float* __restrict__ u_g,
                 const float* __restrict__ v_g, const float* __restrict__ proj,
                 float* __restrict__ soft_raw) {
    __shared__ __align__(16) unsigned short AsH[L_][32];
    __shared__ __align__(16) unsigned short AsL[L_][32];
    __shared__ __align__(16) unsigned short BsH[128][32];  // [t][s]
    __shared__ __align__(16) unsigned short BsL[128][32];

    const int tid = threadIdx.x, lane = tid & 31, wave = tid >> 5;
    const int wm = wave & 3, wn = wave >> 2;
    const int n0 = blockIdx.x * 128, b = blockIdx.y;
    const int am  = lane & 15;
    const int akb = (lane >> 4) << 3;
    const int bn  = lane & 15;
    const int bkb = (lane >> 4) << 3;

    v8f acc[4] = {};

    for (int s0 = 0; s0 < S_; s0 += 32) {
        for (int idx = tid; idx < 64 * 32; idx += 256) {          // P tile (transposed)
            int sp = idx >> 6, l = idx & 63;
            float P = u_g[b * S_ + s0 + sp]
                    * Kmat[(size_t)(b * S_ + s0 + sp) * L_ + l]
                    * v_g[b * L_ + l];
            unsigned short h, lo; split_bf16(P, h, lo);
            AsH[l][sp] = h; AsL[l][sp] = lo;
        }
        for (int idx = tid; idx < 32 * 128; idx += 256) {          // src_proj tile
            int r = idx >> 7, c = idx & 127;
            unsigned short h, lo;
            split_bf16(proj[(size_t)(b * S_ + s0 + r) * T_ + (n0 + c)], h, lo);
            BsH[c][r] = h; BsL[c][r] = lo;
        }
        __syncthreads();
        if (s0 + 32 < S_)
            __builtin_prefetch(&proj[(size_t)(b * S_ + s0 + 32 + (tid & 7)) * T_ + n0], 0, 1);
        const unsigned short* arH = &AsH[wm * 16 + am][0];
        const unsigned short* arL = &AsL[wm * 16 + am][0];
        v16bf aH = ld_frag(arH + akb, arH + akb + 16);
        v16bf aL = ld_frag(arL + akb, arL + akb + 16);
        #pragma unroll
        for (int f = 0; f < 4; ++f) {
            const unsigned short* brH = &BsH[wn * 64 + f * 16 + bn][0];
            const unsigned short* brL = &BsL[wn * 64 + f * 16 + bn][0];
            v16bf bH = ld_frag(brH + bkb, brH + bkb + 16);
            v16bf bL = ld_frag(brL + bkb, brL + bkb + 16);
            acc[f] = wmma_bf16x3(aH, aL, bH, bL, acc[f]);
        }
        __syncthreads();
    }
    #pragma unroll
    for (int f = 0; f < 4; ++f) {
        int col = n0 + wn * 64 + f * 16 + (lane & 15);
        #pragma unroll
        for (int v = 0; v < 8; ++v) {
            int row = wm * 16 + v + ((lane >> 4) << 3);  // l index
            soft_raw[(size_t)(b * L_ + row) * T_ + col] = acc[f][v];
        }
    }
}

// ---------- K5: normalize soft, RMS, output scale, s statistics ----------
__global__ __launch_bounds__(256)
void k_finalize(const float* __restrict__ soft_raw, const float* __restrict__ colsum,
                const float* __restrict__ out_scale_p, float* __restrict__ out,
                float* __restrict__ scal) {
    int bl = blockIdx.x, tid = threadIdx.x;
    float inv = 1.f / (colsum[bl] + 1e-8f);
    const float* rin = soft_raw + (size_t)bl * T_;
    float* rout = out + (size_t)bl * T_;
    float vals[8]; float sq = 0.f;
    #pragma unroll
    for (int i = 0; i < 8; ++i) {
        float x = rin[tid + i * 256] * inv;
        vals[i] = x; sq += x * x;
    }
    __shared__ float red[256];
    __shared__ float smul;
    red[tid] = sq; __syncthreads();
    for (int off = 128; off; off >>= 1) { if (tid < off) red[tid] += red[tid + off]; __syncthreads(); }
    if (!tid) smul = out_scale_p[0] / sqrtf(red[0] * (1.f / (float)T_) + 1e-8f);
    __syncthreads();
    float m = smul, ss = 0.f, s2 = 0.f;
    #pragma unroll
    for (int i = 0; i < 8; ++i) {
        float y = vals[i] * m;
        rout[tid + i * 256] = y;
        ss += y; s2 += y * y;
    }
    red[tid] = ss; __syncthreads();
    for (int off = 128; off; off >>= 1) { if (tid < off) red[tid] += red[tid + off]; __syncthreads(); }
    if (!tid) atomicAdd(&scal[3], red[0]);
    __syncthreads();
    red[tid] = s2; __syncthreads();
    for (int off = 128; off; off >>= 1) { if (tid < off) red[tid] += red[tid + off]; __syncthreads(); }
    if (!tid) atomicAdd(&scal[4], red[0]);
}

__global__ void k_scalars(const float* __restrict__ scal, float* __restrict__ tail) {
    if (threadIdx.x | blockIdx.x) return;
    const float Np = (float)(B_ * S_ * L_);
    const float Ns = (float)(B_ * L_ * T_);
    tail[0] = scal[0] / (float)B_;                          // ot_cost
    float pm = scal[1] / Np;
    float pv = (scal[2] - scal[1] * pm) / (Np - 1.f);
    tail[1] = sqrtf(fmaxf(pv, 0.f));                        // p_std (ddof=1)
    float sm = scal[3] / Ns;
    float sv = (scal[4] - scal[3] * sm) / (Ns - 1.f);
    tail[2] = fmaxf(sv, 0.f);                               // s_var (ddof=1)
}

extern "C" void kernel_launch(void* const* d_in, const int* in_sizes, int n_in,
                              void* d_out, int out_size, void* d_ws, size_t ws_size,
                              hipStream_t stream) {
    (void)in_sizes; (void)n_in; (void)out_size; (void)ws_size;
    const float* src_hidden = (const float*)d_in[0];
    const int*   src_mask   = (const int*)  d_in[1];
    const float* W          = (const float*)d_in[2];
    const float* b_lin      = (const float*)d_in[3];
    const float* anchors    = (const float*)d_in[4];
    const float* cost_scale = (const float*)d_in[5];
    const float* out_scale  = (const float*)d_in[6];
    float* out = (float*)d_out;
    float* ws  = (float*)d_ws;

    float* src_proj = ws;
    float* Cmat     = src_proj + (size_t)M_ * T_;
    float* Kmat     = Cmat + (size_t)M_ * L_;
    float* soft_raw = Kmat + (size_t)M_ * L_;
    float* scale    = soft_raw + (size_t)B_ * L_ * T_;
    float* anchS    = scale + T_;
    float* anchSq   = anchS + (size_t)L_ * T_;
    float* avec     = anchSq + L_;
    float* uvec     = avec + (size_t)B_ * S_;
    float* vvec     = uvec + (size_t)B_ * S_;
    float* colsum   = vvec + B_ * L_;
    float* scal     = colsum + B_ * L_;

    k_scale_zero<<<dim3(T_ / 256), 256, 0, stream>>>(cost_scale, scale, scal);
    k_anchor    <<<dim3(L_),       256, 0, stream>>>(anchors, scale, anchS, anchSq);
    k_avec      <<<dim3(B_),       256, 0, stream>>>(src_mask, avec);
    k_proj_gemm <<<dim3(M_ / PG_BM, T_ / PG_BN), 256, 0, stream>>>(src_hidden, W, b_lin, src_proj);
    k_cost      <<<dim3(M_ / CK_BM), 256, 0, stream>>>(src_proj, scale, anchS, anchSq, Cmat, Kmat);
    k_sinkhorn  <<<dim3(B_),       256, 0, stream>>>(Kmat, avec, uvec, vvec);
    k_pstats    <<<dim3(B_ * L_),  256, 0, stream>>>(Kmat, Cmat, uvec, vvec, colsum, scal);
    k_soft_gemm <<<dim3(T_ / 128, B_), 256, 0, stream>>>(Kmat, uvec, vvec, src_proj, soft_raw);
    k_finalize  <<<dim3(B_ * L_),  256, 0, stream>>>(soft_raw, colsum, out_scale, out, scal);
    k_scalars   <<<1, 1, 0, stream>>>(scal, out + (size_t)B_ * L_ * T_);
}
